// FactorizedQuantizer_781684048561
// MI455X (gfx1250) — matrix operand order repaired and checked by reading, not compiled, not run on hardware
//
#include <hip/hip_runtime.h>
#include <hip/hip_bf16.h>

typedef __attribute__((ext_vector_type(16))) __bf16 bf16x16;
typedef __attribute__((ext_vector_type(8)))  __bf16 bf16x8;
typedef __attribute__((ext_vector_type(8)))  float  v8f;

constexpr int kD     = 512;     // feature dim
constexpr int kNE    = 8192;    // codebook entries
constexpr int kNROWS = 32768;   // B*S latent vectors
constexpr float kEPS = 1e-12f;

// ---------------------------------------------------------------------------
// Kernel 1/2: row-normalize fp32 -> bf16 (one wave32 per row, 16 elems/lane)
// ---------------------------------------------------------------------------
__global__ __launch_bounds__(256) void normalize_to_bf16_kernel(
    const float* __restrict__ x, __bf16* __restrict__ out, int nrows) {
  const int lane = threadIdx.x & 31;
  const int wave = threadIdx.x >> 5;
  const int row  = blockIdx.x * 8 + wave;
  if (row >= nrows) return;

  const float4* xr = (const float4*)(x + (size_t)row * kD);
  float4 v[4];
  float s = 0.f;
#pragma unroll
  for (int i = 0; i < 4; ++i) {
    v[i] = xr[lane * 4 + i];
    s += v[i].x * v[i].x + v[i].y * v[i].y + v[i].z * v[i].z + v[i].w * v[i].w;
  }
#pragma unroll
  for (int ofs = 16; ofs >= 1; ofs >>= 1) s += __shfl_xor(s, ofs, 32);
  const float inv = 1.0f / fmaxf(sqrtf(s), kEPS);

  float f[16];
#pragma unroll
  for (int i = 0; i < 4; ++i) {
    f[4 * i + 0] = v[i].x; f[4 * i + 1] = v[i].y;
    f[4 * i + 2] = v[i].z; f[4 * i + 3] = v[i].w;
  }
  bf16x16 o;
#pragma unroll
  for (int i = 0; i < 16; ++i) o[i] = (__bf16)(f[i] * inv);
  *(bf16x16*)(out + (size_t)row * kD + lane * 16) = o;
}

// ---------------------------------------------------------------------------
// Kernel: zero counts + sumsq accumulator
// ---------------------------------------------------------------------------
__global__ void init_ws_kernel(unsigned* __restrict__ counts, float* __restrict__ sumsq) {
  const int i = blockIdx.x * blockDim.x + threadIdx.x;
  if (i < kNE) counts[i] = 0u;
  if (i == 0) *sumsq = 0.0f;
}

// ---------------------------------------------------------------------------
// Kernel 3: fused bf16 WMMA GEMM (zn @ embn^T) + per-row running argmax.
// Block = 256 threads (8 waves), each wave owns a 16-row M tile (block: 128
// rows). Flattened 1024-stage loop over (64 n-groups x 16 k-chunks).
// B-panels (128x32 bf16) are TRIPLE-buffered in LDS and fetched with async
// global->LDS DMA two stages ahead; one barrier per stage both publishes the
// incoming panel and fences buffer reuse. All 8 B fragments are batch-loaded
// into distinct registers before the WMMA chain so the in-order DS returns
// can be consumed with partial waits instead of 8 full dscnt-0 stalls.
// ---------------------------------------------------------------------------
__global__ __launch_bounds__(256) void vq_wmma_argmax_kernel(
    const __bf16* __restrict__ zn, const __bf16* __restrict__ embn,
    int* __restrict__ idx_out) {
  constexpr int LSTRIDE = 40;   // bf16 elems/row: 80B stride, 16B-aligned, bank-clean
  constexpr int kStages = (kNE / 128) * (kD / 32);  // 1024
  __shared__ __bf16 Bs[3][128 * LSTRIDE];

  const int lane = threadIdx.x & 31;
  const int wave = threadIdx.x >> 5;
  const int m0   = blockIdx.x * 128 + wave * 16;
  const int l15  = lane & 15;
  const int hi   = lane >> 4;                 // lane half
  const int mrow = m0 + l15;                  // A row for this lane
  const int aoff = hi ? 8 : 0;                // A frag K sub-offset
  const int boff = hi ? 8 : 0;                // B frag K sub-offset

  // cooperative B staging: thread t DMAs 32B of one emb row (2 x b128)
  const int srow  = threadIdx.x >> 1;
  const int shalf = (threadIdx.x & 1) * 16;
  // flat->LDS: addr[31:0] is the wave-relative LDS byte offset (ISA 10.2)
  const unsigned ldsDstBase =
      (unsigned)(uintptr_t)(&Bs[0][0]) + (unsigned)((srow * LSTRIDE + shalf) * 2);

  auto issue_stage = [&](int s) {
    const int n0 = (s >> 4) * 128;
    const int k0 = (s & 15) * 32;
    const __bf16* src = embn + (size_t)(n0 + srow) * kD + k0 + shalf;
    const unsigned dst = ldsDstBase + (unsigned)(s % 3) * (unsigned)sizeof(Bs[0]);
    // INST_OFFSET is added to BOTH the LDS and global addresses (ISA 10.x),
    // so offset:16 stages the second 16B of this thread's 32B slice.
    asm volatile(
        "global_load_async_to_lds_b128 %0, %1, off\n\t"
        "global_load_async_to_lds_b128 %0, %1, off offset:16"
        :
        : "v"(dst), "v"(src)
        : "memory");
  };

  float best_val[8];
  int   best_idx[8];
#pragma unroll
  for (int r = 0; r < 8; ++r) { best_val[r] = -3.4e38f; best_idx[r] = 0; }

  const v8f vzero = {};
  v8f c[8] = {};  // 8 n-tiles of f32 accumulators

  issue_stage(0);
  issue_stage(1);
  for (int s = 0; s < kStages; ++s) {
    // A fragment first: its global-load latency hides under the wait+barrier.
    // (CDNA5 16-bit A layout: lane halves split K {0-7,16-23}/{8-15,24-31})
    const int k0 = (s & 15) * 32;
    const __bf16* ap = zn + (size_t)mrow * kD + k0 + aoff;
    const bf16x8 alo = *(const bf16x8*)(ap);
    const bf16x8 ahi = *(const bf16x8*)(ap + 16);
    const bf16x16 a = __builtin_shufflevector(alo, ahi,
        0, 1, 2, 3, 4, 5, 6, 7, 8, 9, 10, 11, 12, 13, 14, 15);

    // async loads complete in order: leaving the 2 newest (stage s+1) in
    // flight guarantees stage s's panel has landed in LDS
    if (s + 1 < kStages) {
      asm volatile("s_wait_asynccnt 0x2" ::: "memory");
    } else {
      asm volatile("s_wait_asynccnt 0x0" ::: "memory");
    }
    __syncthreads();  // publish current panel; fence reuse of buffer (s+2)%3
    if (s + 2 < kStages) issue_stage(s + 2);

    const __bf16* Bcur = &Bs[s % 3][0];
    // batch-load all 8 B fragments (16 ds_load_b128 back-to-back), then run
    // the WMMA chain: in-order DS completion lets waits be partial
    bf16x16 b[8];
#pragma unroll
    for (int j = 0; j < 8; ++j) {
      const __bf16* bp = Bcur + (j * 16 + l15) * LSTRIDE + boff;
      const bf16x8 blo = *(const bf16x8*)(bp);
      const bf16x8 bhi = *(const bf16x8*)(bp + 16);
      b[j] = __builtin_shufflevector(blo, bhi,
          0, 1, 2, 3, 4, 5, 6, 7, 8, 9, 10, 11, 12, 13, 14, 15);
    }
#pragma unroll
    for (int j = 0; j < 8; ++j) {
      c[j] = __builtin_amdgcn_wmma_f32_16x16x32_bf16(
          false, a, false, b[j], (short)0, c[j], false, false);
    }

    if ((s & 15) == 15) {  // full K accumulated for this n-group
      const int n0 = (s >> 4) * 128;
#pragma unroll
      for (int j = 0; j < 8; ++j) {
        const int col = n0 + j * 16 + l15;
#pragma unroll
        for (int r = 0; r < 8; ++r) {
          const float v = c[j][r];
          if (v > best_val[r]) { best_val[r] = v; best_idx[r] = col; }
        }
        c[j] = vzero;
      }
    }
  }

  // cross-lane reduce within 16-lane halves (C row M=r lives on lanes 0-15,
  // row M=r+8 on lanes 16-31); tie-break on lowest index like jnp.argmin
#pragma unroll
  for (int r = 0; r < 8; ++r) {
    float v  = best_val[r];
    int   bi = best_idx[r];
    for (int ofs = 8; ofs >= 1; ofs >>= 1) {
      const float ov = __shfl_xor(v, ofs, 32);
      const int   oi = __shfl_xor(bi, ofs, 32);
      if (ov > v || (ov == v && oi < bi)) { v = ov; bi = oi; }
    }
    if (lane == 0)       idx_out[m0 + r]     = bi;
    else if (lane == 16) idx_out[m0 + 8 + r] = bi;
  }
}

// ---------------------------------------------------------------------------
// Kernel 4: gather z_q = emb[idx], output normalize(z_q), accumulate
// sum((zq_n - z_n)^2), usage histogram, and idx-as-float. One wave per row.
// ---------------------------------------------------------------------------
__global__ __launch_bounds__(256) void gather_loss_kernel(
    const float* __restrict__ z, const float* __restrict__ emb,
    const int* __restrict__ idx, float* __restrict__ out_zq,
    float* __restrict__ out_idx, float* __restrict__ sumsq,
    unsigned* __restrict__ counts) {
  const int lane = threadIdx.x & 31;
  const int wave = threadIdx.x >> 5;
  const int row  = blockIdx.x * 8 + wave;
  if (row >= kNROWS) return;
  const int id = idx[row];

  const float4* zr = (const float4*)(z + (size_t)row * kD);
  const float4* qr = (const float4*)(emb + (size_t)id * kD);
  float4 vz[4], vq[4];
  float sz = 0.f, sq = 0.f;
#pragma unroll
  for (int i = 0; i < 4; ++i) {
    vz[i] = zr[lane * 4 + i];
    vq[i] = qr[lane * 4 + i];
    sz += vz[i].x * vz[i].x + vz[i].y * vz[i].y + vz[i].z * vz[i].z + vz[i].w * vz[i].w;
    sq += vq[i].x * vq[i].x + vq[i].y * vq[i].y + vq[i].z * vq[i].z + vq[i].w * vq[i].w;
  }
#pragma unroll
  for (int ofs = 16; ofs >= 1; ofs >>= 1) {
    sz += __shfl_xor(sz, ofs, 32);
    sq += __shfl_xor(sq, ofs, 32);
  }
  const float iz = 1.0f / fmaxf(sqrtf(sz), kEPS);
  const float iq = 1.0f / fmaxf(sqrtf(sq), kEPS);

  float acc = 0.f;
  float4* outp = (float4*)(out_zq + (size_t)row * kD);
#pragma unroll
  for (int i = 0; i < 4; ++i) {
    float4 qn;
    qn.x = vq[i].x * iq; qn.y = vq[i].y * iq;
    qn.z = vq[i].z * iq; qn.w = vq[i].w * iq;
    const float dx = qn.x - vz[i].x * iz, dy = qn.y - vz[i].y * iz;
    const float dz = qn.z - vz[i].z * iz, dw = qn.w - vz[i].w * iz;
    acc += dx * dx + dy * dy + dz * dz + dw * dw;
    outp[lane * 4 + i] = qn;
  }
#pragma unroll
  for (int ofs = 16; ofs >= 1; ofs >>= 1) acc += __shfl_xor(acc, ofs, 32);
  if (lane == 0) {
    atomicAdd(sumsq, acc);
    atomicAdd(&counts[id], 1u);
    out_idx[row] = (float)id;
  }
}

// ---------------------------------------------------------------------------
// Kernel 5: finalize loss and perplexity
// ---------------------------------------------------------------------------
__global__ __launch_bounds__(256) void finalize_kernel(
    const unsigned* __restrict__ counts, const float* __restrict__ sumsq,
    float* __restrict__ out_loss, float* __restrict__ out_perp) {
  __shared__ float red[256];
  float h = 0.f;
  for (int i = threadIdx.x; i < kNE; i += 256) {
    const float p = (float)counts[i] * (1.0f / (float)kNROWS);
    h += p * logf(p + 1e-10f);
  }
  red[threadIdx.x] = h;
  __syncthreads();
  for (int s = 128; s > 0; s >>= 1) {
    if (threadIdx.x < s) red[threadIdx.x] += red[threadIdx.x + s];
    __syncthreads();
  }
  if (threadIdx.x == 0) {
    *out_perp = expf(-red[0]);
    // loss = mean((zq_n - z_n)^2) + BETA * mean(same) = 1.25 * sumsq / (N*D)
    *out_loss = 1.25f * (*sumsq) / (float)((size_t)kNROWS * kD);
  }
}

// ---------------------------------------------------------------------------
extern "C" void kernel_launch(void* const* d_in, const int* in_sizes, int n_in,
                              void* d_out, int out_size, void* d_ws, size_t ws_size,
                              hipStream_t stream) {
  const float* z   = (const float*)d_in[0];  // [16,2048,512]
  const float* emb = (const float*)d_in[1];  // [8192,512]

  char* ws = (char*)d_ws;
  __bf16*   zn     = (__bf16*)ws;                                  // 32 MB
  __bf16*   embn   = (__bf16*)(ws + (size_t)33554432);             // 8 MB
  int*      idx    = (int*)(ws + (size_t)33554432 + 8388608);      // 128 KB
  unsigned* counts = (unsigned*)(ws + (size_t)33554432 + 8388608 + 131072); // 32 KB
  float*    sumsq  = (float*)(ws + (size_t)33554432 + 8388608 + 131072 + 32768);

  float* out      = (float*)d_out;
  float* out_loss = out;                                     // [1]
  float* out_zq   = out + 1;                                 // [16*2048*512]
  float* out_idx  = out + 1 + (size_t)kNROWS * kD;           // [32768]
  float* out_perp = out + 1 + (size_t)kNROWS * kD + kNROWS;  // [1]

  normalize_to_bf16_kernel<<<kNROWS / 8, 256, 0, stream>>>(z, zn, kNROWS);
  normalize_to_bf16_kernel<<<kNE / 8, 256, 0, stream>>>(emb, embn, kNE);
  init_ws_kernel<<<(kNE + 255) / 256, 256, 0, stream>>>(counts, sumsq);
  vq_wmma_argmax_kernel<<<kNROWS / 128, 256, 0, stream>>>(zn, embn, idx);
  gather_loss_kernel<<<kNROWS / 8, 256, 0, stream>>>(z, emb, idx, out_zq, out_idx,
                                                     sumsq, counts);
  finalize_kernel<<<1, 256, 0, stream>>>(counts, sumsq, out_loss, out_perp);
}
